// RAFT343used_61443802137088
// MI455X (gfx1250) — matrix operand order repaired and checked by reading, not compiled, not run on hardware
//
#include <hip/hip_runtime.h>
#include <hip/hip_bf16.h>
#include <stdint.h>

// ---------------- problem constants (N=1) ----------------
#define HW    7168      // 56*128
#define CCH   256       // feature channels
#define H0    56
#define W0    128
#define NFEAT 324       // 4 levels * 81 offsets
#define KPAD  352       // NFEAT padded to multiple of 32
#define NMASK 576       // 64*9 mask channels
#define NITER_MAX 8

typedef __attribute__((ext_vector_type(16))) __bf16 v16bf;
typedef __attribute__((ext_vector_type(8)))  float  v8f;

// ---------------- helpers ----------------
__device__ __forceinline__ uint16_t f32_to_bf16u(float f) {
    uint32_t u = __float_as_uint(f);
    uint32_t r = (u + 0x7FFFu + ((u >> 16) & 1u)) >> 16;   // round-to-nearest-even
    return (uint16_t)r;
}

struct FragAB { union { v16bf v; uint4 q[2]; }; };

// A-matrix 16x32 bf16 fragment (ISA 7.12.2): lane m (0..15), h = lane>>4.
// elements 0..7  = K[k0 + h*8 .. +7], elements 8..15 = K[k0 + 16 + h*8 .. +7]
__device__ __forceinline__ void load_fragA(FragAB& f, const uint16_t* base, int ld,
                                           int m, int h, int k0) {
    const uint16_t* r = base + (size_t)m * ld + k0 + h * 8;
    f.q[0] = *(const uint4*)(r);
    f.q[1] = *(const uint4*)(r + 16);
}
// B-matrix 32x16 bf16 fragment: lane n, elements 0..7 = K[k0 + h*8 .. +7], 8..15 = K[k0 + 16 + h*8 .. +7]
// (B stored transposed: Bt[n][k], same row-major-K layout as A)
__device__ __forceinline__ void load_fragB(FragAB& f, const uint16_t* base, int ld,
                                           int n, int h, int k0) {
    const uint16_t* r = base + (size_t)n * ld + k0 + h * 8;
    f.q[0] = *(const uint4*)(r);
    f.q[1] = *(const uint4*)(r + 16);
}

// One wave computes a 64x64 f32 tile of D = A * Bt^T (both [rows][K] bf16, row-major).
__device__ __forceinline__ void wave_gemm_64x64(const uint16_t* __restrict__ A, int lda,
                                                const uint16_t* __restrict__ B, int ldb,
                                                float* __restrict__ D, int ldd,
                                                int kTiles, float scale,
                                                int p0, int q0) {
    const int lane = threadIdx.x & 31;
    const int mloc = lane & 15;
    const int h    = lane >> 4;

    v8f acc[4][4];
#pragma unroll
    for (int i = 0; i < 4; ++i)
#pragma unroll
        for (int j = 0; j < 4; ++j)
#pragma unroll
            for (int e = 0; e < 8; ++e) acc[i][j][e] = 0.0f;

    for (int kt = 0; kt < kTiles; ++kt) {
        const int k0 = kt * 32;
        FragAB a[4], b[4];
#pragma unroll
        for (int i = 0; i < 4; ++i) load_fragA(a[i], A, lda, p0 + i * 16 + mloc, h, k0);
#pragma unroll
        for (int j = 0; j < 4; ++j) load_fragB(b[j], B, ldb, q0 + j * 16 + mloc, h, k0);

        if (kt + 1 < kTiles) {  // lowers to global_prefetch_b8
            __builtin_prefetch(A + (size_t)(p0 + mloc) * lda + k0 + 32, 0, 3);
            __builtin_prefetch(B + (size_t)(q0 + mloc) * ldb + k0 + 32, 0, 3);
        }

#pragma unroll
        for (int i = 0; i < 4; ++i)
#pragma unroll
            for (int j = 0; j < 4; ++j)
                acc[i][j] = __builtin_amdgcn_wmma_f32_16x16x32_bf16(
                    false, a[i].v, false, b[j].v, (short)0, acc[i][j], false, false);
    }

    // C/D layout: VGPR e, lanes 0-15 -> (M=e, N=lane); lanes 16-31 -> (M=8+e, N=lane-16)
#pragma unroll
    for (int i = 0; i < 4; ++i)
#pragma unroll
        for (int j = 0; j < 4; ++j)
#pragma unroll
            for (int e = 0; e < 8; ++e) {
                int m = p0 + i * 16 + h * 8 + e;
                int n = q0 + j * 16 + mloc;
                D[(size_t)m * ldd + n] = acc[i][j][e] * scale;
            }
}

// ---------------- kernel 1: f32 [C,HW] -> bf16 [HW,C] (transpose + convert) ----------------
__global__ void transpose_bf16_kernel(const float* __restrict__ src, uint16_t* __restrict__ dst) {
    int idx = blockIdx.x * blockDim.x + threadIdx.x;     // over HW*C, exact grid
    int p = idx / CCH, c = idx % CCH;
    dst[idx] = f32_to_bf16u(src[c * HW + p]);
}

// ---------------- kernel 2: W_mask f32 [576,324] -> bf16 [576,352], *0.25 folded ----------------
__global__ void wmask_prep_kernel(const float* __restrict__ Wm, uint16_t* __restrict__ dst) {
    int idx = blockIdx.x * blockDim.x + threadIdx.x;
    if (idx >= NMASK * KPAD) return;
    int o = idx / KPAD, k = idx % KPAD;
    float v = (k < NFEAT) ? 0.25f * Wm[o * NFEAT + k] : 0.0f;
    dst[idx] = f32_to_bf16u(v);
}

// ---------------- kernel 3: coords1 = coords0 + flow_init ----------------
__global__ void coords_init_kernel(const float* __restrict__ flow, float* __restrict__ coords) {
    int p = blockIdx.x * blockDim.x + threadIdx.x;       // exact: 28*256 = HW
    coords[p]      = (float)(p % W0) + flow[p];          // x
    coords[HW + p] = (float)(p / W0) + flow[HW + p];     // y
}

// ---------------- kernel 4: corr0 = f1t * f2t^T / 16 (WMMA GEMM, M=N=7168, K=256) ----------------
__global__ __launch_bounds__(256) void corr_gemm_kernel(const uint16_t* __restrict__ f1t,
                                                        const uint16_t* __restrict__ f2t,
                                                        float* __restrict__ corr0) {
    int wave = threadIdx.x >> 5;                         // 8 waves: 2 x 4 tiles of 64x64
    int p0 = blockIdx.y * 128 + (wave >> 2) * 64;
    int q0 = blockIdx.x * 256 + (wave & 3) * 64;
    wave_gemm_64x64(f1t, CCH, f2t, CCH, corr0, HW, CCH / 32, 0.0625f, p0, q0);
}

// ---------------- kernel 5: 2x2 average pooling over [HW, Hs, Ws] ----------------
__global__ void avgpool_kernel(const float* __restrict__ src, float* __restrict__ dst,
                               int Hs, int Ws) {
    int Hd = Hs >> 1, Wd = Ws >> 1;
    int idx = blockIdx.x * blockDim.x + threadIdx.x;     // exact grids
    int wd = idx % Wd; int t = idx / Wd;
    int hd = t % Hd;   int b = t / Hd;
    const float* s = src + (size_t)b * Hs * Ws;
    float v = 0.25f * (s[(2 * hd) * Ws + 2 * wd]     + s[(2 * hd) * Ws + 2 * wd + 1] +
                       s[(2 * hd + 1) * Ws + 2 * wd] + s[(2 * hd + 1) * Ws + 2 * wd + 1]);
    dst[idx] = v;
}

// ---------------- kernel 6: multi-scale lookup + delta + coords update ----------------
__device__ __forceinline__ float corr_tap(const float* __restrict__ img, int Hl, int Wl,
                                          int yi, int xi) {
    if (xi < 0 || xi >= Wl || yi < 0 || yi >= Hl) return 0.0f;
    return img[yi * Wl + xi];
}

__global__ __launch_bounds__(256) void lookup_kernel(const float* __restrict__ corr0,
                                                     const float* __restrict__ corr1,
                                                     const float* __restrict__ corr2,
                                                     const float* __restrict__ corr3,
                                                     const float* __restrict__ Wc,
                                                     float* __restrict__ coords,
                                                     uint16_t* __restrict__ feat,
                                                     const int* __restrict__ d_iters, int it) {
    if (it >= *d_iters) return;                          // uniform branch
    __shared__ float sWc[2 * NFEAT];
    for (int i = threadIdx.x; i < 2 * NFEAT; i += blockDim.x) sWc[i] = Wc[i];
    __syncthreads();

    int p = blockIdx.x * blockDim.x + threadIdx.x;       // exact: 28*256 = HW
    float cx = coords[p], cy = coords[HW + p];

    const float* lv[4] = {corr0, corr1, corr2, corr3};
    const int    lh[4] = {56, 28, 14, 7};
    const int    lw[4] = {128, 64, 32, 16};

    float d0 = 0.0f, d1 = 0.0f;
    int f = 0;
    uint16_t* frow = feat + (size_t)p * KPAD;

#pragma unroll
    for (int l = 0; l < 4; ++l) {
        const int Hl = lh[l], Wl = lw[l];
        const float* img = lv[l] + (size_t)p * Hl * Wl;
        const float inv_s = 1.0f / (float)(1 << l);
        const float bx = cx * inv_s, by = cy * inv_s;
        for (int dy = -4; dy <= 4; ++dy) {
            for (int dx = -4; dx <= 4; ++dx) {
                float xs = bx + (float)dx, ys = by + (float)dy;
                float x0 = floorf(xs), y0 = floorf(ys);
                float wx = xs - x0, wy = ys - y0;
                int xi = (int)x0, yi = (int)y0;
                float v00 = corr_tap(img, Hl, Wl, yi,     xi);
                float v01 = corr_tap(img, Hl, Wl, yi,     xi + 1);
                float v10 = corr_tap(img, Hl, Wl, yi + 1, xi);
                float v11 = corr_tap(img, Hl, Wl, yi + 1, xi + 1);
                float val = (1.0f - wx) * (1.0f - wy) * v00 + wx * (1.0f - wy) * v01 +
                            (1.0f - wx) * wy * v10 + wx * wy * v11;
                d0 += sWc[f] * val;
                d1 += sWc[NFEAT + f] * val;
                frow[f] = f32_to_bf16u(val);
                ++f;
            }
        }
    }
    for (; f < KPAD; ++f) frow[f] = 0;                   // zero K padding

    coords[p]      = cx + d0;
    coords[HW + p] = cy + d1;
}

// ---------------- kernel 7: mask logits = Wmask_bf16 * feat^T (WMMA, M=576,N=7168,K=352) ----
__global__ __launch_bounds__(256) void mask_gemm_kernel(const uint16_t* __restrict__ Wmb,
                                                        const uint16_t* __restrict__ feat,
                                                        float* __restrict__ mlogit) {
    int wave = threadIdx.x >> 5;                         // 8 waves: 1 x 8 tiles of 64x64
    int p0 = blockIdx.y * 64;                            // mask channel (576 = 9*64)
    int q0 = blockIdx.x * 512 + wave * 64;               // pixel (7168 = 14*512)
    wave_gemm_64x64(Wmb, KPAD, feat, KPAD, mlogit, HW, KPAD / 32, 1.0f, p0, q0);
}

// ---------------- kernel 8: softmax-9 convex upsample -> [2, 448, 1024] ----------------
__global__ void upsample_kernel(const float* __restrict__ coords,
                                const float* __restrict__ mlogit,
                                float* __restrict__ out) {
    int t = blockIdx.x * blockDim.x + threadIdx.x;       // exact: HW*64
    int p = t % HW;                                      // pixel inner -> coalesced reads
    int r = t / HW;                                      // 0..63 subpixel
    int ry = r >> 3, rx = r & 7;
    int h = p / W0, w = p % W0;

    float m[9]; float mx = -3.0e38f;
#pragma unroll
    for (int k = 0; k < 9; ++k) {
        m[k] = mlogit[(size_t)(k * 64 + r) * HW + p];
        mx = fmaxf(mx, m[k]);
    }
    float sum = 0.0f;
#pragma unroll
    for (int k = 0; k < 9; ++k) { m[k] = __expf(m[k] - mx); sum += m[k]; }
    float inv = 1.0f / sum;

#pragma unroll
    for (int c = 0; c < 2; ++c) {
        float acc = 0.0f;
#pragma unroll
        for (int k = 0; k < 9; ++k) {
            int ky = k / 3, kx = k % 3;
            int hh = h + ky - 1, ww = w + kx - 1;
            float fl = 0.0f;
            if (hh >= 0 && hh < H0 && ww >= 0 && ww < W0) {
                int pp = hh * W0 + ww;
                float base = (c == 0) ? (float)ww : (float)hh;
                fl = 8.0f * (coords[c * HW + pp] - base);
            }
            acc += m[k] * inv * fl;
        }
        out[(size_t)c * (8 * H0 * 8 * W0) + (size_t)(h * 8 + ry) * (8 * W0) + (w * 8 + rx)] = acc;
    }
}

// ---------------- workspace layout (all 256B aligned) ----------------
static constexpr size_t SZ_F1T    = (size_t)HW * CCH * 2;          // bf16 [HW,256]
static constexpr size_t SZ_F2T    = (size_t)HW * CCH * 2;
static constexpr size_t SZ_CORR0  = (size_t)HW * HW * 4;
static constexpr size_t SZ_CORR1  = (size_t)HW * (HW / 4) * 4;
static constexpr size_t SZ_CORR2  = (size_t)HW * (HW / 16) * 4;
static constexpr size_t SZ_CORR3  = (size_t)HW * (HW / 64) * 4;
static constexpr size_t SZ_FEAT   = (size_t)HW * KPAD * 2;         // bf16 [HW,352]
static constexpr size_t SZ_WMASK  = (size_t)NMASK * KPAD * 2;      // bf16 [576,352]
static constexpr size_t SZ_COORDS = (size_t)2 * HW * 4;
static constexpr size_t OFF_F1T    = 0;
static constexpr size_t OFF_F2T    = OFF_F1T + SZ_F1T;
static constexpr size_t OFF_CORR0  = OFF_F2T + SZ_F2T;
static constexpr size_t OFF_CORR1  = OFF_CORR0 + SZ_CORR0;
static constexpr size_t OFF_CORR2  = OFF_CORR1 + SZ_CORR1;
static constexpr size_t OFF_CORR3  = OFF_CORR2 + SZ_CORR2;
static constexpr size_t OFF_FEAT   = OFF_CORR3 + SZ_CORR3;
static constexpr size_t OFF_WMASK  = OFF_FEAT + SZ_FEAT;
static constexpr size_t OFF_COORDS = OFF_WMASK + SZ_WMASK;
static constexpr size_t OFF_MLOGIT = OFF_COORDS + SZ_COORDS;       // f32 [576,HW]

extern "C" void kernel_launch(void* const* d_in, const int* in_sizes, int n_in,
                              void* d_out, int out_size, void* d_ws, size_t ws_size,
                              hipStream_t stream) {
    (void)in_sizes; (void)n_in; (void)out_size; (void)ws_size;

    const float* fmap1 = (const float*)d_in[0];
    const float* fmap2 = (const float*)d_in[1];
    const float* flow  = (const float*)d_in[2];
    const float* Wc    = (const float*)d_in[3];
    const float* Wm    = (const float*)d_in[4];
    const int*   it_p  = (const int*)d_in[5];

    char* ws = (char*)d_ws;
    uint16_t* f1t    = (uint16_t*)(ws + OFF_F1T);
    uint16_t* f2t    = (uint16_t*)(ws + OFF_F2T);
    float*    corr0  = (float*)(ws + OFF_CORR0);
    float*    corr1  = (float*)(ws + OFF_CORR1);
    float*    corr2  = (float*)(ws + OFF_CORR2);
    float*    corr3  = (float*)(ws + OFF_CORR3);
    uint16_t* feat   = (uint16_t*)(ws + OFF_FEAT);
    uint16_t* wmaskb = (uint16_t*)(ws + OFF_WMASK);
    float*    coords = (float*)(ws + OFF_COORDS);
    float*    mlogit = (float*)(ws + OFF_MLOGIT);
    float*    out    = (float*)d_out;

    // 1) convert/transpose feature maps to bf16 [HW, C]
    transpose_bf16_kernel<<<HW * CCH / 256, 256, 0, stream>>>(fmap1, f1t);
    transpose_bf16_kernel<<<HW * CCH / 256, 256, 0, stream>>>(fmap2, f2t);
    // 2) W_mask -> bf16 padded (0.25 folded in)
    wmask_prep_kernel<<<(NMASK * KPAD + 255) / 256, 256, 0, stream>>>(Wm, wmaskb);
    // 3) coords1 = coords0 + flow_init
    coords_init_kernel<<<HW / 256, 256, 0, stream>>>(flow, coords);
    // 4) correlation GEMM (WMMA bf16): 7168 x 7168 x 256
    corr_gemm_kernel<<<dim3(HW / 256, HW / 128), 256, 0, stream>>>(f1t, f2t, corr0);
    // 5) pyramid levels 1..3
    avgpool_kernel<<<(HW * 28 * 64) / 256, 256, 0, stream>>>(corr0, corr1, 56, 128);
    avgpool_kernel<<<(HW * 14 * 32) / 256, 256, 0, stream>>>(corr1, corr2, 28, 64);
    avgpool_kernel<<<(HW * 7 * 16) / 256, 256, 0, stream>>>(corr2, corr3, 14, 32);
    // 6) iterative lookup + coords update (guarded by device-side iters)
    for (int it = 0; it < NITER_MAX; ++it)
        lookup_kernel<<<HW / 256, 256, 0, stream>>>(corr0, corr1, corr2, corr3,
                                                    Wc, coords, feat, it_p, it);
    // 7) mask GEMM (WMMA bf16): 576 x 7168 x 352
    mask_gemm_kernel<<<dim3(HW / 512, NMASK / 64), 256, 0, stream>>>(wmaskb, feat, mlogit);
    // 8) convex upsampling
    upsample_kernel<<<(HW * 64) / 256, 256, 0, stream>>>(coords, mlogit, out);
}
